// LSTMQA_89936615178705
// MI455X (gfx1250) — compile-verified, matrix operand order checked by reading, not run on hardware
//
#include <hip/hip_runtime.h>
#include <hip/hip_bf16.h>

// ---------------------------------------------------------------------------
// BiDAF-style LSTM QA forward for MI455X (gfx1250, wave32, WMMA).
// All GEMMs use v_wmma_f32_16x16x32_f16 (f16 inputs, f32 accumulate).
// Big projections use a 1x4-tile register-blocked GEMM (A-fragment reuse x4,
// ~26 FLOP per L2 byte) with global_prefetch of the next W K-slice.
// LSTM recurrences run as single-workgroup persistent scans with h/c in LDS.
// ---------------------------------------------------------------------------

typedef _Float16 v16h __attribute__((ext_vector_type(16)));
typedef float    v8f  __attribute__((ext_vector_type(8)));

#define NEGINF (-1.0e9f)

// problem dims
#define BATCH 8
#define CLEN  400
#define QLEN  48
#define EMB   300
#define EPAD  320      // E padded to multiple of 32 for WMMA K
#define HID   256
#define HOUT  512      // 2*HID
#define G4    1024     // 4*HID (gates)
#define GDIM  2048     // 4*HOUT (G feature dim)
#define QPAD  64       // QLEN padded to multiple of 32

// ---------------------------------------------------------------------------
// WMMA fragment helpers (ISA 7.12.2 wave32 16-bit layouts)
// A 16x32 (row-major source): lane = {row=lane&15, khalf=(lane>>4)*8},
//   VGPR halves hold K = khalf..khalf+7 and 16+khalf..16+khalf+7.
// B 32x16 from a row-major N x K matrix (i.e. B = W^T): identical pattern with
//   "row" meaning the output column. Both are two contiguous 16B loads/lane.
// ---------------------------------------------------------------------------
__device__ __forceinline__ v16h load_frag_rm(const _Float16* base, int ld, int kk) {
  int lane = threadIdx.x & 31;
  int row  = lane & 15;
  int kb   = (lane >> 4) << 3;            // 0 or 8
  const _Float16* p = base + (size_t)row * ld + kk + kb;
  v16h v;
#pragma unroll
  for (int i = 0; i < 8; ++i) v[i] = p[i];
#pragma unroll
  for (int i = 0; i < 8; ++i) v[8 + i] = p[16 + i];
  return v;
}

// B fragment from a row-major K x N matrix (true NN GEMM), strided loads.
__device__ __forceinline__ v16h load_frag_nn(const _Float16* base, int ld, int kk) {
  int lane = threadIdx.x & 31;
  int col  = lane & 15;
  int kb   = (lane >> 4) << 3;
  v16h v;
#pragma unroll
  for (int i = 0; i < 8; ++i) v[i]     = base[(size_t)(kk + kb + i) * ld + col];
#pragma unroll
  for (int i = 0; i < 8; ++i) v[8 + i] = base[(size_t)(kk + 16 + kb + i) * ld + col];
  return v;
}

__device__ __forceinline__ v8f v8zero() {
  v8f a;
#pragma unroll
  for (int i = 0; i < 8; ++i) a[i] = 0.f;
  return a;
}

__device__ __forceinline__ v8f wmma_f16(v16h a, v16h b, v8f c) {
  // D = A x B + C, 16x16x32, f32 accumulate
  return __builtin_amdgcn_wmma_f32_16x16x32_f16(false, a, false, b, (short)0, c,
                                                false, false);
}

__device__ __forceinline__ float sigmoidf(float x) { return 1.f / (1.f + expf(-x)); }

// ---------------------------------------------------------------------------
// fp32 -> f16 conversion with K padding (weights, activations)
// ---------------------------------------------------------------------------
__global__ void cvt_f16_pad(const float* __restrict__ src, _Float16* __restrict__ dst,
                            int rows, int cols, int ldd) {
  long i = (long)blockIdx.x * blockDim.x + threadIdx.x;
  long total = (long)rows * ldd;
  if (i >= total) return;
  int r = (int)(i / ldd);
  int c = (int)(i - (long)r * ldd);
  dst[i] = (c < cols) ? (_Float16)src[(size_t)r * cols + c] : (_Float16)0.f;
}

// embedding gather directly into padded f16 rows
__global__ void embed_gather_f16(const int* __restrict__ tok,
                                 const float* __restrict__ emb,
                                 _Float16* __restrict__ dst, int ntok) {
  long i = (long)blockIdx.x * blockDim.x + threadIdx.x;
  long total = (long)ntok * EPAD;
  if (i >= total) return;
  int t = (int)(i / EPAD);
  int k = (int)(i - (long)t * EPAD);
  dst[i] = (k < EMB) ? (_Float16)emb[(size_t)tok[t] * EMB + k] : (_Float16)0.f;
}

// ---------------------------------------------------------------------------
// Register-blocked WMMA GEMM, NT:  Y[M,N] = A[M,K] * W[N,K]^T (+ bias[N])
// One wave computes a 16x64 strip (4 N-tiles), reusing the A fragment 4x.
// Requires M%16==0, N%64==0, K%32==0. batch in blockIdx.y.
// ---------------------------------------------------------------------------
__global__ void wmma_gemm_nt4(const _Float16* __restrict__ A, size_t sA,
                              const _Float16* __restrict__ W, size_t sW,
                              const float* __restrict__ bias,
                              float* __restrict__ Y, size_t sY,
                              int M, int N, int K, int lda, int ldw, int ldy) {
  int wpb = blockDim.x >> 5;
  int wid = blockIdx.x * wpb + (threadIdx.x >> 5);
  int Mt = M >> 4, Ng = N >> 6;               // N groups of 64
  if (wid >= Mt * Ng) return;                 // uniform per wave; EXEC stays all-1s
  int mt = wid / Ng, ng = wid - mt * Ng;
  const _Float16* Ab = A + (size_t)blockIdx.y * sA + (size_t)(mt * 16) * lda;
  const _Float16* Wb = W + (size_t)blockIdx.y * sW + (size_t)(ng * 64) * ldw;
  float* Yb = Y + (size_t)blockIdx.y * sY;

  v8f acc[4];
#pragma unroll
  for (int g = 0; g < 4; ++g) acc[g] = v8zero();

  for (int kk = 0; kk < K; kk += 32) {
    v16h a = load_frag_rm(Ab, lda, kk);       // reused by all 4 N-tiles
    if (kk + 64 < K) {                        // prefetch next K-slice of W
      __builtin_prefetch(Wb + (size_t)((threadIdx.x & 31) & 15) * ldw + kk + 64, 0, 1);
    }
#pragma unroll
    for (int g = 0; g < 4; ++g) {
      v16h b = load_frag_rm(Wb + (size_t)(g * 16) * ldw, ldw, kk);
      acc[g] = wmma_f16(a, b, acc[g]);
    }
  }
  int lane = threadIdx.x & 31;
  int col  = lane & 15;
  int r0   = (lane >> 4) << 3;                // C/D: rows r0..r0+7 in 8 VGPRs
#pragma unroll
  for (int g = 0; g < 4; ++g) {
    const int nc = ng * 64 + g * 16 + col;
    const float bv = bias ? bias[nc] : 0.f;
#pragma unroll
    for (int i = 0; i < 8; ++i)
      Yb[(size_t)(mt * 16 + r0 + i) * ldy + nc] = acc[g][i] + bv;
  }
}

// Single-tile NT variant for small N (S einsum, N=48).
__global__ void wmma_gemm_nt(const _Float16* __restrict__ A, size_t sA,
                             const _Float16* __restrict__ W, size_t sW,
                             const float* __restrict__ bias,
                             float* __restrict__ Y, size_t sY,
                             int M, int N, int K, int lda, int ldw, int ldy) {
  int wpb = blockDim.x >> 5;
  int wid = blockIdx.x * wpb + (threadIdx.x >> 5);
  int Mt = M >> 4, Nt = N >> 4;
  if (wid >= Mt * Nt) return;
  int mt = wid / Nt, nt = wid - mt * Nt;
  const _Float16* Ab = A + (size_t)blockIdx.y * sA + (size_t)(mt * 16) * lda;
  const _Float16* Wb = W + (size_t)blockIdx.y * sW + (size_t)(nt * 16) * ldw;
  float* Yb = Y + (size_t)blockIdx.y * sY;

  v8f acc = v8zero();
  for (int kk = 0; kk < K; kk += 32) {
    v16h a = load_frag_rm(Ab, lda, kk);
    v16h b = load_frag_rm(Wb, ldw, kk);
    acc = wmma_f16(a, b, acc);
  }
  int lane = threadIdx.x & 31;
  int col  = lane & 15;
  int r0   = (lane >> 4) << 3;
  float bv = bias ? bias[nt * 16 + col] : 0.f;
#pragma unroll
  for (int i = 0; i < 8; ++i)
    Yb[(size_t)(mt * 16 + r0 + i) * ldy + nt * 16 + col] = acc[i] + bv;
}

// NN variant: Y[M,N] = A[M,K] * Bm[K,N]   (used for softmax(S) @ Q)
__global__ void wmma_gemm_nn(const _Float16* __restrict__ A, size_t sA,
                             const _Float16* __restrict__ Bm, size_t sB,
                             float* __restrict__ Y, size_t sY,
                             int M, int N, int K, int lda, int ldb, int ldy) {
  int wpb = blockDim.x >> 5;
  int wid = blockIdx.x * wpb + (threadIdx.x >> 5);
  int Mt = M >> 4, Nt = N >> 4;
  if (wid >= Mt * Nt) return;
  int mt = wid / Nt, nt = wid - mt * Nt;
  const _Float16* Ab = A + (size_t)blockIdx.y * sA + (size_t)(mt * 16) * lda;
  const _Float16* Bb = Bm + (size_t)blockIdx.y * sB + (nt * 16);
  float* Yb = Y + (size_t)blockIdx.y * sY;

  v8f acc = v8zero();
  for (int kk = 0; kk < K; kk += 32) {
    v16h a = load_frag_rm(Ab, lda, kk);
    v16h b = load_frag_nn(Bb, ldb, kk);
    acc = wmma_f16(a, b, acc);
  }
  int lane = threadIdx.x & 31;
  int col  = lane & 15;
  int r0   = (lane >> 4) << 3;
#pragma unroll
  for (int i = 0; i < 8; ++i)
    Yb[(size_t)(mt * 16 + r0 + i) * ldy + nt * 16 + col] = acc[i];
}

// ---------------------------------------------------------------------------
// Persistent LSTM scan. grid.x = 2 (0=fwd, 1=bwd), block = 1024 (32 waves).
// Each step: g[8,1024] = pre[:,t] + h @ Whh^T via WMMA (M=8 padded to 16,
// K=256, N=1024 -> 64 N-tiles, 2 per wave sharing the A fragment), then
// fp32 gate math. h/c/h16 live in LDS; Whh (f16, 512KB) is L2-resident.
// ---------------------------------------------------------------------------
__global__ void lstm_scan(const float* __restrict__ pre0, const float* __restrict__ pre1,
                          const _Float16* __restrict__ whh0, const _Float16* __restrict__ whh1,
                          const int* __restrict__ mask,   // (B,T) or nullptr => 1.0
                          float* __restrict__ out,        // (B,T,512), fwd->[0:256), bwd->[256:512)
                          int T) {
  __shared__ float    g_s[8 * 1024];
  __shared__ float    h_s[2048];
  __shared__ float    c_s[2048];
  __shared__ _Float16 h16[16 * 256];   // rows 8..15 stay zero (M padding)

  const int dir = blockIdx.x;
  const float*    pre = dir ? pre1 : pre0;
  const _Float16* whh = dir ? whh1 : whh0;
  const int doff = dir ? HID : 0;

  const int tid = threadIdx.x;
  h_s[tid] = 0.f; h_s[tid + 1024] = 0.f;
  c_s[tid] = 0.f; c_s[tid + 1024] = 0.f;
#pragma unroll
  for (int i = 0; i < 4; ++i) h16[tid + i * 1024] = (_Float16)0.f;
  __syncthreads();

  const int wid  = tid >> 5;
  const int lane = tid & 31;
  const int col  = lane & 15;
  const int nc0  = (wid * 2) * 16;
  const int nc1  = nc0 + 16;

  for (int s = 0; s < T; ++s) {
    const int t = dir ? (T - 1 - s) : s;

    // ---- recurrent matmul: g = h @ Whh^T ------------------------------
    v8f acc0 = v8zero(), acc1 = v8zero();
    for (int kk = 0; kk < HID; kk += 32) {
      v16h a  = load_frag_rm(h16, HID, kk);
      v16h b0 = load_frag_rm(whh + (size_t)nc0 * HID, HID, kk);
      v16h b1 = load_frag_rm(whh + (size_t)nc1 * HID, HID, kk);
      acc0 = wmma_f16(a, b0, acc0);
      acc1 = wmma_f16(a, b1, acc1);
    }
    // rows 0..7 (= batch) come from lanes 0..15; lanes 16..31 hold pad rows
    if (lane < 16) {
#pragma unroll
      for (int i = 0; i < 8; ++i) {
        const size_t prow = ((size_t)i * T + t) * G4;
        g_s[i * 1024 + nc0 + col] = acc0[i] + pre[prow + nc0 + col];
        g_s[i * 1024 + nc1 + col] = acc1[i] + pre[prow + nc1 + col];
      }
    }
    __syncthreads();

    // ---- gates (fp32 VALU), 2 (b,j) pairs per thread -------------------
#pragma unroll
    for (int pp = 0; pp < 2; ++pp) {
      const int p  = tid + pp * 1024;       // 0..2047
      const int b  = p >> 8;
      const int j  = p & 255;
      const float gi = sigmoidf(g_s[b * 1024 + j]);
      const float gf = sigmoidf(g_s[b * 1024 + 256 + j]);
      const float gg = tanhf   (g_s[b * 1024 + 512 + j]);
      const float go = sigmoidf(g_s[b * 1024 + 768 + j]);
      const float c_old = c_s[p];
      const float h_old = h_s[p];
      const float c_new = gf * c_old + gi * gg;
      const float h_new = go * tanhf(c_new);
      const float mt = mask ? (float)mask[(size_t)b * T + t] : 1.0f;
      const float h2 = mt * h_new + (1.f - mt) * h_old;
      const float c2 = mt * c_new + (1.f - mt) * c_old;
      h_s[p] = h2;
      c_s[p] = c2;
      h16[b * 256 + j] = (_Float16)h2;
      out[((size_t)b * T + t) * HOUT + doff + j] = mt * h_new;
    }
    __syncthreads();
  }
}

// ---------------------------------------------------------------------------
// Attention helpers
// ---------------------------------------------------------------------------
__global__ void row_dot(const float* __restrict__ X, const float* __restrict__ w,
                        float* __restrict__ out, int rows, int K) {
  int r = blockIdx.x * blockDim.x + threadIdx.x;
  if (r >= rows) return;
  const float* x = X + (size_t)r * K;
  float acc = 0.f;
  for (int k = 0; k < K; ++k) acc += x[k] * w[k];
  out[r] = acc;
}

// Y[i] = (f16)(X[i] * w[i & 511])   (C * w3)
__global__ void mul_vec_f16(const float* __restrict__ X, const float* __restrict__ w,
                            _Float16* __restrict__ Y, long n) {
  long i = (long)blockIdx.x * blockDim.x + threadIdx.x;
  if (i >= n) return;
  Y[i] = (_Float16)(X[i] * w[i & (HOUT - 1)]);
}

// Qpad[b, k(0..63), n] = k<48 ? Q[b,k,n] : 0   (K x N operand for NN GEMM)
__global__ void qpack_f16(const float* __restrict__ Q, _Float16* __restrict__ Qpad) {
  long i = (long)blockIdx.x * blockDim.x + threadIdx.x;
  long total = (long)BATCH * QPAD * HOUT;
  if (i >= total) return;
  int n = (int)(i & (HOUT - 1));
  int k = (int)((i >> 9) & (QPAD - 1));
  int b = (int)(i >> 15);
  Qpad[i] = (k < QLEN) ? (_Float16)Q[((size_t)b * QLEN + k) * HOUT + n] : (_Float16)0.f;
}

// one wave per (b,c) row: S = Sraw + c1 + q2, masked; softmax over q; maxS.
__global__ void softmax_q(const float* __restrict__ Sraw, const float* __restrict__ c1,
                          const float* __restrict__ q2, const int* __restrict__ qmask,
                          _Float16* __restrict__ P_h, float* __restrict__ maxS) {
  const int lane = threadIdx.x & 31;
  const int row  = blockIdx.x * (blockDim.x >> 5) + (threadIdx.x >> 5); // 0..3199
  const int b    = row / CLEN;
  const float* Sr = Sraw + (size_t)row * QLEN;
  const float cc = c1[row];
  const int l2 = lane + 32;
  float v0 = (lane < QLEN && qmask[b * QLEN + lane]) ? Sr[lane] + cc + q2[b * QLEN + lane] : NEGINF;
  float v1 = (l2   < QLEN && qmask[b * QLEN + l2])   ? Sr[l2]   + cc + q2[b * QLEN + l2]   : NEGINF;
  float m = fmaxf(v0, v1);
#pragma unroll
  for (int o = 16; o > 0; o >>= 1) m = fmaxf(m, __shfl_xor(m, o, 32));
  float e0 = (lane < QLEN) ? expf(v0 - m) : 0.f;
  float e1 = (l2   < QLEN) ? expf(v1 - m) : 0.f;
  float sum = e0 + e1;
#pragma unroll
  for (int o = 16; o > 0; o >>= 1) sum += __shfl_xor(sum, o, 32);
  const float inv = 1.f / sum;
  _Float16* Pr = P_h + (size_t)row * QPAD;
  Pr[lane] = (_Float16)(e0 * inv);
  Pr[l2]   = (_Float16)((l2 < QLEN) ? e1 * inv : 0.f);
  if (lane == 0) maxS[row] = m;
}

// per batch: B_att = softmax_c(maxS); Bvec[b,h] = sum_c B_att[c] * C[b,c,h]
__global__ void softmax_c_bvec(const float* __restrict__ maxS, const float* __restrict__ C,
                               float* __restrict__ Bvec) {
  const int b = blockIdx.x, tid = threadIdx.x;            // block = 256
  __shared__ float sm[CLEN];
  __shared__ float red[256];
  const float* ms = maxS + (size_t)b * CLEN;
  float m = -3.0e38f;
  for (int c = tid; c < CLEN; c += 256) m = fmaxf(m, ms[c]);
  red[tid] = m; __syncthreads();
  for (int o = 128; o > 0; o >>= 1) { if (tid < o) red[tid] = fmaxf(red[tid], red[tid + o]); __syncthreads(); }
  m = red[0]; __syncthreads();
  float s = 0.f;
  for (int c = tid; c < CLEN; c += 256) { float e = expf(ms[c] - m); sm[c] = e; s += e; }
  red[tid] = s; __syncthreads();
  for (int o = 128; o > 0; o >>= 1) { if (tid < o) red[tid] += red[tid + o]; __syncthreads(); }
  const float inv = 1.f / red[0]; __syncthreads();
  for (int c = tid; c < CLEN; c += 256) sm[c] *= inv;
  __syncthreads();
  for (int h = tid; h < HOUT; h += 256) {
    float acc = 0.f;
    for (int c = 0; c < CLEN; ++c) acc += sm[c] * C[((size_t)b * CLEN + c) * HOUT + h];
    Bvec[(size_t)b * HOUT + h] = acc;
  }
}

// G_h[row, :] = f16{ C, A, C*A, C*Bvec }   (row = b*400+c)
__global__ void build_G(const float* __restrict__ C, const float* __restrict__ A,
                        const float* __restrict__ Bvec, _Float16* __restrict__ G_h) {
  long i = (long)blockIdx.x * blockDim.x + threadIdx.x;
  long total = (long)BATCH * CLEN * HOUT;
  if (i >= total) return;
  const int h   = (int)(i & (HOUT - 1));
  const long row = i >> 9;
  const int b   = (int)(row / CLEN);
  const float c = C[i];
  const float a = A[i];
  _Float16* g = G_h + (size_t)row * GDIM + h;
  g[0]    = (_Float16)c;
  g[512]  = (_Float16)a;
  g[1024] = (_Float16)(c * a);
  g[1536] = (_Float16)(c * Bvec[(size_t)b * HOUT + h]);
}

// output heads: s, e per (b,c); na per b. out = s(3200) ++ e(3200) ++ na(8)
__global__ void final_heads(const float* __restrict__ M,
                            const float* __restrict__ w_s, const float* __restrict__ b_s,
                            const float* __restrict__ w_e, const float* __restrict__ b_e,
                            const float* __restrict__ w_na, const float* __restrict__ b_na,
                            const int* __restrict__ cmask, float* __restrict__ out) {
  const int i = blockIdx.x * blockDim.x + threadIdx.x;
  if (i < BATCH * CLEN) {
    const float* m = M + (size_t)i * HOUT;
    float ss = 0.f, ee = 0.f;
    for (int k = 0; k < HOUT; ++k) { const float v = m[k]; ss += v * w_s[k]; ee += v * w_e[k]; }
    ss += b_s[0]; ee += b_e[0];
    if (cmask[i] == 0) { ss = NEGINF; ee = NEGINF; }
    out[i] = ss;
    out[BATCH * CLEN + i] = ee;
  }
  if (i < BATCH) {
    const float* m = M + (size_t)i * CLEN * HOUT;   // M[b, 0, :]
    float v = 0.f;
    for (int k = 0; k < HOUT; ++k) v += m[k] * w_na[k];
    out[2 * BATCH * CLEN + i] = v + b_na[0];
  }
}

// ---------------------------------------------------------------------------
// host orchestration
// ---------------------------------------------------------------------------
extern "C" void kernel_launch(void* const* d_in, const int* in_sizes, int n_in,
                              void* d_out, int out_size, void* d_ws, size_t ws_size,
                              hipStream_t stream) {
  // ---- inputs (setup_inputs dict flattening) ----
  const int*   ctx      = (const int*)d_in[0];
  const int*   ctx_mask = (const int*)d_in[1];
  const int*   qtok     = (const int*)d_in[2];
  const int*   q_mask   = (const int*)d_in[3];
  const float* emb_c    = (const float*)d_in[4];
  const float* emb_q    = (const float*)d_in[5];
  const float* c_wihf = (const float*)d_in[6],  *c_whhf = (const float*)d_in[7],  *c_bf = (const float*)d_in[8];
  const float* c_wihb = (const float*)d_in[9],  *c_whhb = (const float*)d_in[10], *c_bb = (const float*)d_in[11];
  const float* q_wihf = (const float*)d_in[12], *q_whhf = (const float*)d_in[13], *q_bf = (const float*)d_in[14];
  const float* q_wihb = (const float*)d_in[15], *q_whhb = (const float*)d_in[16], *q_bb = (const float*)d_in[17];
  const float* m_wihf = (const float*)d_in[18], *m_whhf = (const float*)d_in[19], *m_bf = (const float*)d_in[20];
  const float* m_wihb = (const float*)d_in[21], *m_whhb = (const float*)d_in[22], *m_bb = (const float*)d_in[23];
  const float* w_sim  = (const float*)d_in[24];
  const float* w_s = (const float*)d_in[25], *b_s = (const float*)d_in[26];
  const float* w_e = (const float*)d_in[27], *b_e = (const float*)d_in[28];
  const float* w_na = (const float*)d_in[29], *b_na = (const float*)d_in[30];
  float* out = (float*)d_out;

  // ---- workspace carve-up (~81 MB total) ----
  char* wsp = (char*)d_ws;
  size_t cur = 0;
  auto alloc = [&](size_t bytes) -> void* {
    cur = (cur + 255) & ~(size_t)255;
    void* p = wsp + cur;
    cur += bytes;
    return p;
  };
  _Float16* Xc_h   = (_Float16*)alloc((size_t)BATCH * CLEN * EPAD * 2);
  _Float16* Xq_h   = (_Float16*)alloc((size_t)BATCH * QLEN * EPAD * 2);
  _Float16* wihcf  = (_Float16*)alloc((size_t)G4 * EPAD * 2);
  _Float16* wihcb  = (_Float16*)alloc((size_t)G4 * EPAD * 2);
  _Float16* wihqf  = (_Float16*)alloc((size_t)G4 * EPAD * 2);
  _Float16* wihqb  = (_Float16*)alloc((size_t)G4 * EPAD * 2);
  _Float16* wihmf  = (_Float16*)alloc((size_t)G4 * GDIM * 2);
  _Float16* wihmb  = (_Float16*)alloc((size_t)G4 * GDIM * 2);
  _Float16* whhcf  = (_Float16*)alloc((size_t)G4 * HID * 2);
  _Float16* whhcb  = (_Float16*)alloc((size_t)G4 * HID * 2);
  _Float16* whhqf  = (_Float16*)alloc((size_t)G4 * HID * 2);
  _Float16* whhqb  = (_Float16*)alloc((size_t)G4 * HID * 2);
  _Float16* whhmf  = (_Float16*)alloc((size_t)G4 * HID * 2);
  _Float16* whhmb  = (_Float16*)alloc((size_t)G4 * HID * 2);
  float* preF  = (float*)alloc((size_t)BATCH * CLEN * G4 * 4);   // reused ctx/q/mod
  float* preB  = (float*)alloc((size_t)BATCH * CLEN * G4 * 4);
  float* Cbuf  = (float*)alloc((size_t)BATCH * CLEN * HOUT * 4);
  float* Qbuf  = (float*)alloc((size_t)BATCH * QLEN * HOUT * 4);
  float* c1    = (float*)alloc((size_t)BATCH * CLEN * 4);
  float* q2    = (float*)alloc((size_t)BATCH * QLEN * 4);
  _Float16* CW3_h = (_Float16*)alloc((size_t)BATCH * CLEN * HOUT * 2);
  _Float16* Qh    = (_Float16*)alloc((size_t)BATCH * QLEN * HOUT * 2);
  _Float16* Qpad  = (_Float16*)alloc((size_t)BATCH * QPAD * HOUT * 2);
  float* Sraw  = (float*)alloc((size_t)BATCH * CLEN * QLEN * 4);
  float* maxS  = (float*)alloc((size_t)BATCH * CLEN * 4);
  _Float16* P_h = (_Float16*)alloc((size_t)BATCH * CLEN * QPAD * 2);
  float* Bvec  = (float*)alloc((size_t)BATCH * HOUT * 4);
  float* Abuf  = (float*)alloc((size_t)BATCH * CLEN * HOUT * 4);
  _Float16* G_h = (_Float16*)alloc((size_t)BATCH * CLEN * GDIM * 2);
  float* Mbuf  = (float*)alloc((size_t)BATCH * CLEN * HOUT * 4);

  auto cdiv = [](long a, long b) { return (int)((a + b - 1) / b); };
  auto cvt = [&](const float* s, _Float16* d, int rows, int cols, int ldd) {
    cvt_f16_pad<<<cdiv((long)rows * ldd, 256), 256, 0, stream>>>(s, d, rows, cols, ldd);
  };
  // register-blocked projection GEMM (N % 64 == 0)
  auto gemm4 = [&](const _Float16* A, const _Float16* W, const float* bias, float* Y,
                   int M, int N, int K) {
    dim3 grid(cdiv((long)(M / 16) * (N / 64), 8), 1);
    wmma_gemm_nt4<<<grid, 256, 0, stream>>>(A, 0, W, 0, bias, Y, 0, M, N, K, K, K, N);
  };

  // ---- weight conversion (fp32 -> f16, K-padded) ----
  cvt(c_wihf, wihcf, G4, EMB, EPAD);  cvt(c_wihb, wihcb, G4, EMB, EPAD);
  cvt(q_wihf, wihqf, G4, EMB, EPAD);  cvt(q_wihb, wihqb, G4, EMB, EPAD);
  cvt(m_wihf, wihmf, G4, GDIM, GDIM); cvt(m_wihb, wihmb, G4, GDIM, GDIM);
  cvt(c_whhf, whhcf, G4, HID, HID);   cvt(c_whhb, whhcb, G4, HID, HID);
  cvt(q_whhf, whhqf, G4, HID, HID);   cvt(q_whhb, whhqb, G4, HID, HID);
  cvt(m_whhf, whhmf, G4, HID, HID);   cvt(m_whhb, whhmb, G4, HID, HID);

  // ---- embeddings ----
  embed_gather_f16<<<cdiv((long)BATCH * CLEN * EPAD, 256), 256, 0, stream>>>(ctx, emb_c, Xc_h, BATCH * CLEN);
  embed_gather_f16<<<cdiv((long)BATCH * QLEN * EPAD, 256), 256, 0, stream>>>(qtok, emb_q, Xq_h, BATCH * QLEN);

  // ---- context BiLSTM ----
  gemm4(Xc_h, wihcf, c_bf, preF, BATCH * CLEN, G4, EPAD);
  gemm4(Xc_h, wihcb, c_bb, preB, BATCH * CLEN, G4, EPAD);
  lstm_scan<<<2, 1024, 0, stream>>>(preF, preB, whhcf, whhcb, ctx_mask, Cbuf, CLEN);

  // ---- query BiLSTM ----
  gemm4(Xq_h, wihqf, q_bf, preF, BATCH * QLEN, G4, EPAD);
  gemm4(Xq_h, wihqb, q_bb, preB, BATCH * QLEN, G4, EPAD);
  lstm_scan<<<2, 1024, 0, stream>>>(preF, preB, whhqf, whhqb, q_mask, Qbuf, QLEN);

  // ---- attention ----
  row_dot<<<cdiv(BATCH * CLEN, 256), 256, 0, stream>>>(Cbuf, w_sim, c1, BATCH * CLEN, HOUT);
  row_dot<<<cdiv(BATCH * QLEN, 256), 256, 0, stream>>>(Qbuf, w_sim + HOUT, q2, BATCH * QLEN, HOUT);
  mul_vec_f16<<<cdiv((long)BATCH * CLEN * HOUT, 256), 256, 0, stream>>>(Cbuf, w_sim + 2 * HOUT, CW3_h,
                                                                        (long)BATCH * CLEN * HOUT);
  cvt(Qbuf, Qh, BATCH * QLEN, HOUT, HOUT);
  qpack_f16<<<cdiv((long)BATCH * QPAD * HOUT, 256), 256, 0, stream>>>(Qbuf, Qpad);

  // Sraw[b] = (C*w3)[b] (400x512) x Q[b] (48x512)^T   (N=48 -> single-tile NT)
  {
    dim3 grid(cdiv((long)(CLEN / 16) * (QLEN / 16), 8), BATCH);
    wmma_gemm_nt<<<grid, 256, 0, stream>>>(CW3_h, (size_t)CLEN * HOUT, Qh, (size_t)QLEN * HOUT,
                                           nullptr, Sraw, (size_t)CLEN * QLEN,
                                           CLEN, QLEN, HOUT, HOUT, HOUT, QLEN);
  }
  softmax_q<<<CLEN, 256, 0, stream>>>(Sraw, c1, q2, q_mask, P_h, maxS);
  softmax_c_bvec<<<BATCH, 256, 0, stream>>>(maxS, Cbuf, Bvec);

  // A[b] = P[b] (400x64) x Qpad[b] (64x512)
  {
    dim3 grid(cdiv((long)(CLEN / 16) * (HOUT / 16), 8), BATCH);
    wmma_gemm_nn<<<grid, 256, 0, stream>>>(P_h, (size_t)CLEN * QPAD, Qpad, (size_t)QPAD * HOUT,
                                           Abuf, (size_t)CLEN * HOUT,
                                           CLEN, HOUT, QPAD, QPAD, HOUT, HOUT);
  }
  build_G<<<cdiv((long)BATCH * CLEN * HOUT, 256), 256, 0, stream>>>(Cbuf, Abuf, Bvec, G_h);

  // ---- modeling BiLSTM ----
  gemm4(G_h, wihmf, m_bf, preF, BATCH * CLEN, G4, GDIM);
  gemm4(G_h, wihmb, m_bb, preB, BATCH * CLEN, G4, GDIM);
  lstm_scan<<<2, 1024, 0, stream>>>(preF, preB, whhmf, whhmb, nullptr, Mbuf, CLEN);

  // ---- output heads ----
  final_heads<<<cdiv(BATCH * CLEN, 256), 256, 0, stream>>>(Mbuf, w_s, b_s, w_e, b_e, w_na, b_na,
                                                           ctx_mask, out);
}
